// Attention_21139829031374
// MI455X (gfx1250) — matrix operand order repaired and checked by reading, not compiled, not run on hardware
//
#include <hip/hip_runtime.h>
#include <stdint.h>

// Problem constants (from reference): N=4, C=256, CI=128, H=W=64
#define NB   4
#define CC   256
#define CCI  128
#define HWD  4096
#define KT   64     // kv tile size in the flash loop

typedef __attribute__((ext_vector_type(16))) __bf16 v16bf;
typedef __attribute__((ext_vector_type(8)))  float  v8f;
typedef __attribute__((__vector_size__(4 * sizeof(int)))) int v4i;

union AF { v16bf v; __bf16 e[16]; unsigned u[8]; };
union CF { v8f   v; float  f[8]; };

#define AS1 __attribute__((address_space(1)))
#define AS3 __attribute__((address_space(3)))

__device__ __forceinline__ v8f wmma_bf16(v16bf a, v16bf b, v8f c) {
  // D = A(16x32 bf16) * B(32x16 bf16) + C(16x16 f32)
  return __builtin_amdgcn_wmma_f32_16x16x32_bf16(
      /*neg_a=*/false, a, /*neg_b=*/false, b,
      /*c_mod=*/(short)0, c, /*reuse_a=*/false, /*reuse_b=*/false);
}

// 16B async Global->LDS copy (CDNA5 GLOBAL_LOAD_ASYNC_TO_LDS_B128, ASYNCcnt).
__device__ __forceinline__ void async_cp_b128(const void* g, void* l) {
#if __has_builtin(__builtin_amdgcn_global_load_async_to_lds_b128)
  __builtin_amdgcn_global_load_async_to_lds_b128(
      (AS1 v4i*)(unsigned long long)(uintptr_t)g,
      (AS3 v4i*)(unsigned)(uintptr_t)l, 0, 0);
#else
  *(uint4*)l = *(const uint4*)g;   // synchronous fallback
#endif
}

__device__ __forceinline__ void async_wait0() {
#if __has_builtin(__builtin_amdgcn_s_wait_asynccnt)
  __builtin_amdgcn_s_wait_asynccnt(0);
#else
  asm volatile("s_wait_asynccnt 0x0" ::: "memory");
#endif
}

// ---------------------------------------------------------------------------
// Projection GEMM: out[oc, p] = sum_c W[oc,c] * X[b,c,p] + bias[oc]
// X: f32 [C, HW] per batch (1x1 conv == channel matmul), W: f32 [OC, C].
// Output bf16, either [HW, OC] (HWMAJOR) or [OC, HW] — compile-time.
// ---------------------------------------------------------------------------
template <int OC, bool HWMAJOR>
__global__ __launch_bounds__(128) void proj_gemm_kernel(
    const float* __restrict__ Xall, const float* __restrict__ Wm,
    const float* __restrict__ bias, unsigned short* __restrict__ outAll) {
  const int b = blockIdx.z;
  const float* X = Xall + (size_t)b * CC * HWD;
  __bf16* out = (__bf16*)outAll + (size_t)b * (size_t)OC * HWD;

  const int wave = threadIdx.x >> 5;
  const int lane = threadIdx.x & 31;
  const int l16  = lane & 15;
  const int hh   = lane >> 4;
  const int n0   = (blockIdx.x * 4 + wave) * 16;  // HW tile
  const int m0   = blockIdx.y * 16;               // OC tile

  CF acc;
#pragma unroll
  for (int v = 0; v < 8; ++v) acc.f[v] = bias[m0 + v + 8 * hh];

  const int mrow = m0 + l16;
#pragma unroll 1
  for (int k0 = 0; k0 < CC; k0 += 32) {
    AF a, bf;
#pragma unroll
    for (int v = 0; v < 8; ++v) {
      // A layout: lane = row (oc), VGPR v holds K-pair 2v+(v>=4?8:0)+8h
      const int ka = k0 + 2 * v + (v >= 4 ? 8 : 0) + 8 * hh;
      const float2 w2 = *(const float2*)(Wm + (size_t)mrow * CC + ka);
      a.e[2 * v]     = (__bf16)w2.x;
      a.e[2 * v + 1] = (__bf16)w2.y;
      // B layout: lane = col (hw), VGPR v holds K-pair 2v+(v>=4?8:0)+8h
      const int kb = k0 + 2 * v + (v >= 4 ? 8 : 0) + 8 * hh;
      bf.e[2 * v]     = (__bf16)X[(size_t)kb * HWD + n0 + l16];
      bf.e[2 * v + 1] = (__bf16)X[(size_t)(kb + 1) * HWD + n0 + l16];
    }
    acc.v = wmma_bf16(a.v, bf.v, acc.v);
  }

#pragma unroll
  for (int v = 0; v < 8; ++v) {
    const int m = m0 + v + 8 * hh;   // oc
    const int n = n0 + l16;          // hw
    const size_t idx = HWMAJOR ? ((size_t)n * OC + m) : ((size_t)m * HWD + n);
    out[idx] = (__bf16)acc.f[v];
  }
}

// ---------------------------------------------------------------------------
// Fused flash attention with online softmax.
// theta: bf16 [HW, CI]; phiT: bf16 [HW, CI] (kv-major); gT: bf16 [C, HW]
// block = 128 threads (4 waves); each wave owns 16 query rows (block: 64).
// kv streamed in tiles of 64, double-buffered in LDS via async DMA.
// Per tile: 16 QK + 32 PV + 2 l-sum WMMAs. The softmax denominator is
// accumulated as an extra WMMA column (P x ones) -> no sum shuffles, and the
// epilogue reads each row's denominator from the lane's own fragment.
// ---------------------------------------------------------------------------
__global__ __launch_bounds__(128) void flash_attn_kernel(
    const unsigned short* __restrict__ thetaAll,
    const unsigned short* __restrict__ phiTAll,
    const unsigned short* __restrict__ gTAll,
    float* __restrict__ out) {
  __shared__ __bf16 ldsK[2][KT][CCI];  // phi tiles [kv][c]   2 x 16 KB
  __shared__ __bf16 ldsV[2][CC][KT];   // g   tiles [c][kv]   2 x 32 KB
  __shared__ __bf16 ldsP[4][16][KT];   // per-wave P bounce   8 KB

  const int b = blockIdx.y;
  const __bf16* Q  = (const __bf16*)thetaAll + (size_t)b * HWD * CCI;
  const __bf16* Kp = (const __bf16*)phiTAll  + (size_t)b * HWD * CCI;  // [HW][CI]
  const __bf16* Vp = (const __bf16*)gTAll    + (size_t)b * CC * HWD;   // [C][HW]

  const int tid  = threadIdx.x;
  const int wave = tid >> 5;
  const int lane = tid & 31;
  const int l16  = lane & 15;
  const int hh   = lane >> 4;
  const int q0   = blockIdx.x * 64 + wave * 16;
  const float scale = 0.0883883476483184f;  // 1/sqrt(128)

  // Issue async DMA for one kv tile: K 16KB (1024 x b128), V 32KB (2048 x b128)
  auto issue_tile = [&](int kt, int buf) {
    // K tile: 64 kv-rows x 256B (16 chunks each), source rows contiguous
#pragma unroll
    for (int i = 0; i < 8; ++i) {
      const int e = i * 128 + tid;          // 0..1023
      const int r = e >> 4, cpk = e & 15;   // kv row, 16B chunk within 128 ch
      async_cp_b128(Kp + (size_t)(kt + r) * CCI + cpk * 8, &ldsK[buf][r][cpk * 8]);
    }
    // V tile: 256 channel-rows x 128B (8 chunks each)
#pragma unroll
    for (int i = 0; i < 16; ++i) {
      const int e = i * 128 + tid;          // 0..2047
      const int r = e >> 3, cpk = e & 7;    // channel row, 16B chunk of 64 kv
      async_cp_b128(Vp + (size_t)r * HWD + kt + cpk * 8, &ldsV[buf][r][cpk * 8]);
    }
  };

  // Q fragments (A layout), CI=128 in 4 chunks of K=32; pairs are contiguous.
  AF qf[4];
  {
    const int row = q0 + l16;
#pragma unroll
    for (int kc = 0; kc < 4; ++kc)
#pragma unroll
      for (int v = 0; v < 8; ++v) {
        const int kk = kc * 32 + 2 * v + (v >= 4 ? 8 : 0) + 8 * hh;
        qf[kc].u[v] = *(const unsigned*)(Q + (size_t)row * CCI + kk);
      }
  }

  // All-ones B fragment for the denominator column (constant, no loads).
  AF ones;
#pragma unroll
  for (int i = 0; i < 16; ++i) ones.e[i] = (__bf16)1.0f;

  CF o[16], lacc;
#pragma unroll
  for (int t = 0; t < 16; ++t)
#pragma unroll
    for (int v = 0; v < 8; ++v) o[t].f[v] = 0.0f;
#pragma unroll
  for (int v = 0; v < 8; ++v) lacc.f[v] = 0.0f;

  float mrow[8];
#pragma unroll
  for (int v = 0; v < 8; ++v) mrow[v] = -1e30f;

  issue_tile(0, 0);

  const int NT = HWD / KT;
  for (int it = 0; it < NT; ++it) {
    const int cur = it & 1;
    async_wait0();       // this wave's DMA for tile `it` complete
    __syncthreads();     // every wave's share of the tile is in LDS
    if (it + 1 < NT) issue_tile((it + 1) * KT, cur ^ 1);  // overlap next tile

    // ---- S = Q . K^T (four 16x16 D fragments covering 64 kv cols) ----
    CF s[4];
#pragma unroll
    for (int j = 0; j < 4; ++j) {
#pragma unroll
      for (int v = 0; v < 8; ++v) s[j].f[v] = 0.0f;
#pragma unroll
      for (int kc = 0; kc < 4; ++kc) {
        AF bfrag;
#pragma unroll
        for (int v = 0; v < 8; ++v) {
          const int c   = kc * 32 + 2 * v + (v >= 4 ? 8 : 0) + 8 * hh;  // B K index (channel)
          const int kvl = 16 * j + l16;               // B N index (kv)
          bfrag.u[v] = *(const unsigned*)&ldsK[cur][kvl][c];  // pair contiguous
        }
        s[j].v = wmma_bf16(qf[kc].v, bfrag.v, s[j].v);
      }
    }

    // ---- Online softmax: row max (16-lane butterfly), exp, stash P ----
    float alpha[8];
#pragma unroll
    for (int v = 0; v < 8; ++v) {
      float sv[4];
      float mx = -1e30f;
#pragma unroll
      for (int j = 0; j < 4; ++j) {
        sv[j] = s[j].f[v] * scale;
        mx = fmaxf(mx, sv[j]);
      }
#pragma unroll
      for (int off = 1; off < 16; off <<= 1) mx = fmaxf(mx, __shfl_xor(mx, off, 32));
      const float mnew = fmaxf(mrow[v], mx);
      alpha[v] = __expf(mrow[v] - mnew);
      mrow[v] = mnew;
#pragma unroll
      for (int j = 0; j < 4; ++j)
        ldsP[wave][v + 8 * hh][16 * j + l16] = (__bf16)__expf(sv[j] - mnew);
    }
#pragma unroll
    for (int t = 0; t < 16; ++t)
#pragma unroll
      for (int v = 0; v < 8; ++v) o[t].f[v] *= alpha[v];
#pragma unroll
    for (int v = 0; v < 8; ++v) lacc.f[v] *= alpha[v];

    // P as two A fragments, K = 0..31 / 32..63 (same-wave LDS RAW in-order)
    AF pf[2];
#pragma unroll
    for (int q = 0; q < 2; ++q)
#pragma unroll
      for (int v = 0; v < 8; ++v) {
        const int kk = q * 32 + 2 * v + (v >= 4 ? 8 : 0) + 8 * hh;
        pf[q].u[v] = *(const unsigned*)&ldsP[wave][l16][kk];
      }

    // Denominator column: lacc += P . ones
    lacc.v = wmma_bf16(pf[0].v, ones.v, lacc.v);
    lacc.v = wmma_bf16(pf[1].v, ones.v, lacc.v);

    // ---- O += P . V  over 16 column tiles of C=256, 2 K-chunks each ----
#pragma unroll
    for (int t = 0; t < 16; ++t) {
#pragma unroll
      for (int q = 0; q < 2; ++q) {
        AF vfrag;
#pragma unroll
        for (int v = 0; v < 8; ++v) {
          const int kv  = q * 32 + 2 * v + (v >= 4 ? 8 : 0) + 8 * hh;  // B K index (pair contiguous)
          const int col = t * 16 + l16;              // B N index (channel)
          vfrag.u[v] = *(const unsigned*)&ldsV[cur][col][kv];
        }
        o[t].v = wmma_bf16(pf[q].v, vfrag.v, o[t].v);
      }
    }
    __syncthreads();  // all waves done with `cur` before it is refilled
  }

  // ---- Epilogue: divide by denominator, write transposed [C, HW] ----
#pragma unroll
  for (int t = 0; t < 16; ++t)
#pragma unroll
    for (int v = 0; v < 8; ++v) {
      const int q  = q0 + v + 8 * hh;
      const int cc = t * 16 + l16;
      out[((size_t)b * CC + cc) * HWD + q] = o[t].f[v] / lacc.f[v];
    }
}

// ---------------------------------------------------------------------------
extern "C" void kernel_launch(void* const* d_in, const int* in_sizes, int n_in,
                              void* d_out, int out_size, void* d_ws, size_t ws_size,
                              hipStream_t stream) {
  (void)in_sizes; (void)n_in; (void)out_size; (void)ws_size;
  const float* x    = (const float*)d_in[0];
  const float* xref = (const float*)d_in[1];
  const float* Wg   = (const float*)d_in[2];
  const float* bg   = (const float*)d_in[3];
  const float* Wt   = (const float*)d_in[4];
  const float* bt   = (const float*)d_in[5];
  const float* Wp   = (const float*)d_in[6];
  const float* bp   = (const float*)d_in[7];
  float* out = (float*)d_out;

  // bf16 workspace: theta [N][HW][CI], phiT [N][HW][CI], gT [N][C][HW] (~16MB)
  unsigned short* theta = (unsigned short*)d_ws;
  unsigned short* phiT  = theta + (size_t)NB * HWD * CCI;
  unsigned short* gT    = phiT  + (size_t)NB * HWD * CCI;

  dim3 blk(128);
  // theta = Wt . x          -> [HW, CI] per batch
  proj_gemm_kernel<CCI, true><<<dim3(HWD / 64, CCI / 16, NB), blk, 0, stream>>>(x, Wt, bt, theta);
  // phiT  = Wp . x_ref      -> [HW, CI] per batch (kv-major for LDS K tiles)
  proj_gemm_kernel<CCI, true><<<dim3(HWD / 64, CCI / 16, NB), blk, 0, stream>>>(xref, Wp, bp, phiT);
  // gT    = Wg . x_ref      -> [C, HW]  per batch (c-major for LDS V tiles)
  proj_gemm_kernel<CC, false><<<dim3(HWD / 64, CC / 16, NB), blk, 0, stream>>>(xref, Wg, bg, gT);
  // fused softmax(theta.phiT^T/sqrt(CI)) . g -> out [N, C, H, W]
  flash_attn_kernel<<<dim3(HWD / 64, NB), blk, 0, stream>>>(theta, phiT, gT, out);
}
